// TimeLSTMCell0_37254546325730
// MI455X (gfx1250) — compile-verified, matrix operand order checked by reading, not run on hardware
//
#include <hip/hip_runtime.h>
#include <hip/hip_bf16.h>

// ---------------------------------------------------------------------------
// TimeLSTM cell, CDNA5 (gfx1250) WMMA bf16 implementation, v2.
//   - v_wmma_f32_16x16x32_bf16 for both GEMMs
//   - global_load_async_to_lds_b128 staging (ASYNCcnt) + double-buffered LDS
//   - Z staged in bf16 (halves intermediate HBM traffic)
// ---------------------------------------------------------------------------

typedef __attribute__((ext_vector_type(16))) __bf16      bfvec16;
typedef __attribute__((ext_vector_type(8)))  float       fvec8;
typedef __attribute__((ext_vector_type(4)))  unsigned int u32x4;

#define BB   8192
#define FEAT 512
#define UU   1024
#define NZ   5120   // 5*U
#define LDS_STRIDE 40  // 32 K elems + 8 pad (80B rows keep 16B alignment)

__device__ __forceinline__ unsigned short f2bf(float x) {
  unsigned int u = __float_as_uint(x);
  unsigned int r = u + 0x7FFFu + ((u >> 16) & 1u);   // round-to-nearest-even
  return (unsigned short)(r >> 16);
}

__device__ __forceinline__ float bf2f(unsigned short u) {
  return __uint_as_float((unsigned int)u << 16);
}

__device__ __forceinline__ float hsig(float x) {
  return fminf(fmaxf(0.2f * x + 0.5f, 0.0f), 1.0f);
}

// low 32 bits of a generic pointer to LDS == wave-relative LDS byte address
__device__ __forceinline__ unsigned lds_off(const void* p) {
  return (unsigned)(size_t)p;
}

// async DMA: 16 bytes global -> LDS, tracked by ASYNCcnt (no VGPR round-trip)
__device__ __forceinline__ void async_b128(unsigned ldsbyte, const void* gbase,
                                           unsigned gbyteoff) {
  asm volatile("global_load_async_to_lds_b128 %0, %1, %2"
               :
               : "v"(ldsbyte), "v"(gbyteoff),
                 "s"((unsigned long long)(size_t)gbase)
               : "memory");
}

// ---------------- conversion kernels (one-shot, bandwidth-trivial) ---------

__global__ __launch_bounds__(256) void cvt_x(const float* __restrict__ comb,
                                             unsigned short* __restrict__ Xbf) {
  int idx = blockIdx.x * 256 + threadIdx.x;          // over 8192*512
  int b = idx >> 9, f = idx & 511;
  Xbf[idx] = f2bf(comb[(size_t)b * (FEAT + 1) + f]);
}

__global__ __launch_bounds__(256) void cvt_wt(const float* __restrict__ W,
                                              unsigned short* __restrict__ WbfT) {
  int idx = blockIdx.x * 256 + threadIdx.x;          // over 5120*512
  int n = idx >> 9, k = idx & 511;
  WbfT[idx] = f2bf(W[(size_t)k * NZ + n]);           // transpose to N-major
}

__global__ __launch_bounds__(256) void cvt_h(const float* __restrict__ H,
                                             unsigned short* __restrict__ Hbf) {
  int idx = blockIdx.x * 256 + threadIdx.x;          // over 8192*1024
  Hbf[idx] = f2bf(H[idx]);
}

__global__ __launch_bounds__(256) void cvt_rt(const float* __restrict__ R,
                                              unsigned short* __restrict__ RbfT) {
  int idx = blockIdx.x * 256 + threadIdx.x;          // over 4096*1024
  int n = idx >> 10, k = idx & 1023;
  RbfT[idx] = f2bf(R[(size_t)k * (4 * UU) + n]);     // transpose to N-major
}

// ---------------- fragment load from LDS (A and B share the layout) --------
// 16-bit A-matrix 16x32 per ISA 7.12.2: lane<16 holds row M=lane, K runs
// [0..7],[16..23]; lane>=16 holds row M=lane-16, K runs [8..15],[24..31].
__device__ __forceinline__ bfvec16 lds_frag(const unsigned short* base,
                                            int row16, int l16, int half) {
  union { bfvec16 v; u32x4 q[2]; } u;
  const unsigned short* p = base + (row16 + l16) * LDS_STRIDE + half * 8;
  u.q[0] = *(const u32x4*)(p);
  u.q[1] = *(const u32x4*)(p + 16);
  return u.v;
}

// ---------------- GEMM1: Zbf = X @ W + bias --------------------------------
// grid = (NZ/128, B/128), block = 256 (8 waves). Wave (wm,wn) = 4x2 grid,
// each wave computes a 32(M) x 64(N) region = 2x4 wmma accumulators.
__global__ __launch_bounds__(256) void gemm_z(
    const unsigned short* __restrict__ Xbf,
    const unsigned short* __restrict__ WbfT,
    const float* __restrict__ bias,
    unsigned short* __restrict__ Zbf) {
  __shared__ unsigned short As[2][128 * LDS_STRIDE];
  __shared__ unsigned short Bs[2][128 * LDS_STRIDE];

  const int tid  = threadIdx.x;
  const int lane = tid & 31, wave = tid >> 5;
  const int half = lane >> 4, l16 = lane & 15;
  const int wm = wave >> 1, wn = wave & 1;
  const int m0 = blockIdx.y * 128, n0 = blockIdx.x * 128;

  // per-thread chunk assignment: 512 x 16B chunks per tile, 2 per thread
  const int row_c[2] = {(tid + 0)   >> 2, (tid + 256) >> 2};
  const int ko_c[2]  = {(tid & 3) * 8,    (tid & 3) * 8};

  auto issue = [&](int b, int k0) {
#pragma unroll
    for (int c = 0; c < 2; ++c) {
      int row = row_c[c], ko = ko_c[c];
      async_b128(lds_off(&As[b][row * LDS_STRIDE + ko]), Xbf,
                 (unsigned)(((m0 + row) * FEAT + k0 + ko) * 2));
      async_b128(lds_off(&Bs[b][row * LDS_STRIDE + ko]), WbfT,
                 (unsigned)(((n0 + row) * FEAT + k0 + ko) * 2));
    }
  };

  const fvec8 vzero = {0.f, 0.f, 0.f, 0.f, 0.f, 0.f, 0.f, 0.f};
  fvec8 acc[2][4];
#pragma unroll
  for (int i = 0; i < 2; ++i)
#pragma unroll
    for (int j = 0; j < 4; ++j) acc[i][j] = vzero;

  const int NIT = FEAT / 32;  // 16
  int buf = 0;
  issue(0, 0);
  for (int it = 0; it < NIT; ++it) {
    if (it + 1 < NIT) {
      issue(buf ^ 1, (it + 1) * 32);
      asm volatile("s_wait_asynccnt 0x4" ::: "memory");  // cur buffer resident
    } else {
      asm volatile("s_wait_asynccnt 0x0" ::: "memory");
    }
    __syncthreads();

    bfvec16 afr[2], bfr[4];
#pragma unroll
    for (int i = 0; i < 2; ++i)
      afr[i] = lds_frag(As[buf], wm * 32 + i * 16, l16, half);
#pragma unroll
    for (int j = 0; j < 4; ++j)
      bfr[j] = lds_frag(Bs[buf], wn * 64 + j * 16, l16, half);
#pragma unroll
    for (int i = 0; i < 2; ++i)
#pragma unroll
      for (int j = 0; j < 4; ++j)
        acc[i][j] = __builtin_amdgcn_wmma_f32_16x16x32_bf16(
            false, afr[i], false, bfr[j], (short)0, acc[i][j], false, false);
    __syncthreads();   // all waves done reading buf before it is re-filled
    buf ^= 1;
  }

  // epilogue: D element v -> row = v + 8*half, col = l16 (ISA 7.12.2)
#pragma unroll
  for (int i = 0; i < 2; ++i)
#pragma unroll
    for (int j = 0; j < 4; ++j) {
      int col = n0 + wn * 64 + j * 16 + l16;
      float bv = bias[col];
#pragma unroll
      for (int v = 0; v < 8; ++v) {
        int row = m0 + wm * 32 + i * 16 + half * 8 + v;
        Zbf[(size_t)row * NZ + col] = f2bf(acc[i][j][v] + bv);
      }
    }
}

// ---------------- GEMM2 + fused LSTM epilogue ------------------------------
// grid = (U/64, B/128), block = 256.  Wave (wm,wn) = 4x2; each wave owns a
// 32(M) x 32(N) region for ALL FOUR gates -> 4 gates * 2x2 accumulators.
__global__ __launch_bounds__(256) void gemm_rec_epilogue(
    const unsigned short* __restrict__ Hbf,
    const unsigned short* __restrict__ RbfT,   // (4U) x U, N-major
    const unsigned short* __restrict__ Zbf,
    const float* __restrict__ comb,            // for delta_t column
    const float* __restrict__ time_kernel,     // 2048
    const float* __restrict__ c_tm1,
    float* __restrict__ hout) {
  __shared__ unsigned short Hs[2][128 * LDS_STRIDE];
  __shared__ unsigned short Rs[2][256 * LDS_STRIDE];  // 4 gates x 64 N-rows

  const int tid  = threadIdx.x;
  const int lane = tid & 31, wave = tid >> 5;
  const int half = lane >> 4, l16 = lane & 15;
  const int wm = wave >> 1, wn = wave & 1;
  const int m0 = blockIdx.y * 128, n0 = blockIdx.x * 64;

  auto issue = [&](int b, int k0) {
#pragma unroll
    for (int c = 0; c < 2; ++c) {               // H tile: 512 chunks
      int chunk = tid + c * 256;
      int row = chunk >> 2;
      int ko  = (chunk & 3) * 8;
      async_b128(lds_off(&Hs[b][row * LDS_STRIDE + ko]), Hbf,
                 (unsigned)(((m0 + row) * UU + k0 + ko) * 2));
    }
#pragma unroll
    for (int c = 0; c < 4; ++c) {               // R tiles: 1024 chunks
      int chunk = tid + c * 256;
      int r   = chunk >> 2;                     // 0..255 (gate*64 + n)
      int ko  = (chunk & 3) * 8;
      int gate = r >> 6, n = r & 63;
      int grow = gate * UU + n0 + n;            // row in RbfT
      async_b128(lds_off(&Rs[b][r * LDS_STRIDE + ko]), RbfT,
                 (unsigned)((grow * UU + k0 + ko) * 2));
    }
  };

  const fvec8 vzero = {0.f, 0.f, 0.f, 0.f, 0.f, 0.f, 0.f, 0.f};
  fvec8 acc[4][2][2];
#pragma unroll
  for (int g = 0; g < 4; ++g)
#pragma unroll
    for (int i = 0; i < 2; ++i)
#pragma unroll
      for (int j = 0; j < 2; ++j) acc[g][i][j] = vzero;

  const int NIT = UU / 32;  // 32
  int buf = 0;
  issue(0, 0);
  for (int it = 0; it < NIT; ++it) {
    if (it + 1 < NIT) {
      issue(buf ^ 1, (it + 1) * 32);
      asm volatile("s_wait_asynccnt 0x6" ::: "memory");
    } else {
      asm volatile("s_wait_asynccnt 0x0" ::: "memory");
    }
    __syncthreads();

    bfvec16 afr[2];
#pragma unroll
    for (int i = 0; i < 2; ++i)
      afr[i] = lds_frag(Hs[buf], wm * 32 + i * 16, l16, half);
#pragma unroll
    for (int g = 0; g < 4; ++g) {
      bfvec16 bfr[2];
#pragma unroll
      for (int j = 0; j < 2; ++j)
        bfr[j] = lds_frag(Rs[buf], g * 64 + wn * 32 + j * 16, l16, half);
#pragma unroll
      for (int i = 0; i < 2; ++i)
#pragma unroll
        for (int j = 0; j < 2; ++j)
          acc[g][i][j] = __builtin_amdgcn_wmma_f32_16x16x32_bf16(
              false, afr[i], false, bfr[j], (short)0, acc[g][i][j], false, false);
    }
    __syncthreads();
    buf ^= 1;
  }

  // fused LSTM epilogue
#pragma unroll
  for (int i = 0; i < 2; ++i)
#pragma unroll
    for (int j = 0; j < 2; ++j) {
      int n = n0 + wn * 32 + j * 16 + l16;      // 0..U-1
      float tkT = time_kernel[n];
      float tko = time_kernel[UU + n];
#pragma unroll
      for (int v = 0; v < 8; ++v) {
        int r = m0 + wm * 32 + i * 16 + half * 8 + v;
        size_t zb = (size_t)r * NZ;
        float xi = bf2f(Zbf[zb + n]);
        float xf = bf2f(Zbf[zb + UU + n]);
        float xc = bf2f(Zbf[zb + 2 * UU + n]);
        float xo = bf2f(Zbf[zb + 3 * UU + n]);
        float xT = bf2f(Zbf[zb + 4 * UU + n]);
        float delta = comb[(size_t)r * (FEAT + 1) + FEAT];
        float T  = hsig(xT + hsig(delta * tkT));
        float ig = hsig(xi + acc[0][i][j][v]);
        float fg = hsig(xf + acc[1][i][j][v]);
        float cn = fg * c_tm1[(size_t)r * UU + n] +
                   ig * T * tanhf(xc + acc[2][i][j][v]);
        float og = hsig(xo + delta * tko + acc[3][i][j][v]);
        hout[(size_t)r * UU + n] = og * tanhf(cn);
      }
    }
}

// ---------------------------------------------------------------------------

extern "C" void kernel_launch(void* const* d_in, const int* in_sizes, int n_in,
                              void* d_out, int out_size, void* d_ws, size_t ws_size,
                              hipStream_t stream) {
  const float* comb   = (const float*)d_in[0];   // (8192, 513)
  const float* h_tm1  = (const float*)d_in[1];   // (8192, 1024)
  const float* c_tm1  = (const float*)d_in[2];   // (8192, 1024)
  const float* kern   = (const float*)d_in[3];   // (512, 5120)
  const float* tkern  = (const float*)d_in[4];   // (1, 2048)
  const float* rkern  = (const float*)d_in[5];   // (1024, 4096)
  const float* bias   = (const float*)d_in[6];   // (5120,)
  float* hout = (float*)d_out;

  // workspace layout
  char* ws = (char*)d_ws;
  size_t off = 0;
  unsigned short* Zbf  = (unsigned short*)(ws + off); off += (size_t)BB * NZ * 2;   // 84 MB
  unsigned short* Xbf  = (unsigned short*)(ws + off); off += (size_t)BB * FEAT * 2; // 8 MB
  unsigned short* WbfT = (unsigned short*)(ws + off); off += (size_t)NZ * FEAT * 2; // 5 MB
  unsigned short* Hbf  = (unsigned short*)(ws + off); off += (size_t)BB * UU * 2;   // 16 MB
  unsigned short* RbfT = (unsigned short*)(ws + off); off += (size_t)4 * UU * UU * 2;// 8 MB

  cvt_x <<<(BB * FEAT) / 256, 256, 0, stream>>>(comb, Xbf);
  cvt_wt<<<(NZ * FEAT) / 256, 256, 0, stream>>>(kern, WbfT);
  cvt_h <<<(BB * UU) / 256, 256, 0, stream>>>(h_tm1, Hbf);
  cvt_rt<<<(4 * UU * UU) / 256, 256, 0, stream>>>(rkern, RbfT);

  gemm_z<<<dim3(NZ / 128, BB / 128), 256, 0, stream>>>(Xbf, WbfT, bias, Zbf);

  gemm_rec_epilogue<<<dim3(UU / 64, BB / 128), 256, 0, stream>>>(
      Hbf, RbfT, Zbf, comb, tkern, c_tm1, hout);
}
